// HNO_8641474199773
// MI455X (gfx1250) — matrix operand order repaired and checked by reading, not compiled, not run on hardware
//
#include <hip/hip_runtime.h>
#include <stdint.h>

#define HD 128
#define LRELU 0.01f

typedef __attribute__((ext_vector_type(16))) __bf16 v16bf;
typedef __attribute__((ext_vector_type(16))) float  v16f;
typedef __attribute__((ext_vector_type(8)))  float  v8f;

// ---------- graph preprocessing ----------
__global__ void k_deg(float* deg, const int* src, int E) {
    int e = blockIdx.x * blockDim.x + threadIdx.x;
    if (e < E) atomicAdd(deg + src[e], 1.0f);
}

__global__ void k_edgew(float* w, const float* deg, const int* src, const int* dst, int E) {
    int e = blockIdx.x * blockDim.x + threadIdx.x;
    if (e >= E) return;
    float ds = deg[src[e]], dd = deg[dst[e]];
    float a = ds > 0.f ? rsqrtf(fmaxf(ds, 1.f)) : 0.f;
    float b = dd > 0.f ? rsqrtf(fmaxf(dd, 1.f)) : 0.f;
    w[e] = -a * b;
}

// ---------- sparse propagation ----------
__global__ void k_scatter3(float* out, const float* t, const float* w,
                           const int* src, const int* dst, float s, int E) {
    int e = blockIdx.x * blockDim.x + threadIdx.x;
    if (e >= E) return;
    float wv = s * w[e];
    int si = src[e], di = dst[e];
    atomicAdd(out + di * 3 + 0, wv * t[si * 3 + 0]);
    atomicAdd(out + di * 3 + 1, wv * t[si * 3 + 1]);
    atomicAdd(out + di * 3 + 2, wv * t[si * 3 + 2]);
}

__global__ void k_scatter128(float* out, const float* __restrict__ t, const float* __restrict__ w,
                             const int* __restrict__ src, const int* __restrict__ dst, float s, int E) {
    int tid = blockIdx.x * blockDim.x + threadIdx.x;   // E*32 threads, 4 cols each
    if (tid >= E * 32) return;
    int e = tid >> 5;
    int c = (tid & 31) * 4;
    float wv = s * w[e];
    const float4 v = *reinterpret_cast<const float4*>(t + (size_t)src[e] * HD + c);
    float* o = out + (size_t)dst[e] * HD + c;
    atomicAdd(o + 0, wv * v.x);
    atomicAdd(o + 1, wv * v.y);
    atomicAdd(o + 2, wv * v.z);
    atomicAdd(o + 3, wv * v.w);
}

__global__ void k_negcopy(float* out, const float* in, int n) {
    int i = blockIdx.x * blockDim.x + threadIdx.x;
    if (i < n) out[i] = -in[i];
}

// ---------- layer 1 (3 -> 128, scalar; K too small for WMMA) ----------
__global__ void k_dense1(float* out, const float* t0, const float* t1, const float* t2, const float* t3,
                         const float* W, const float* b, int Nn) {
    int i = blockIdx.x * blockDim.x + threadIdx.x;   // n*128 + h
    if (i >= Nn * HD) return;
    int n = i >> 7, h = i & 127;
    const float* T[4] = {t0, t1, t2, t3};
    float acc = b[h];
#pragma unroll
    for (int k = 0; k < 4; ++k) {
        const float* tn = T[k] + n * 3;
        const float* wk = W + k * 3 * HD + h;
        acc += tn[0] * wk[0] + tn[1] * wk[HD] + tn[2] * wk[2 * HD];
    }
    out[i] = acc;
}

// ---------- weight transpose to bf16 [t][col][k] ----------
__global__ void k_twei(__bf16* wt, const float* W) {
    int i = blockIdx.x * blockDim.x + threadIdx.x;   // t*16384 + col*128 + k
    if (i >= 4 * HD * HD) return;
    int t = i >> 14, r = i & 16383, n = r >> 7, k = r & 127;
    wt[i] = (__bf16)W[t * HD * HD + k * HD + n];
}

// ---------- fused Chebyshev dense mix: z = sum_t T_t @ W[t] + b (bf16 WMMA) ----------
__global__ void __launch_bounds__(256)
k_cheb_wmma(float* __restrict__ z,
            const float* __restrict__ t0, const float* __restrict__ t1,
            const float* __restrict__ t2, const float* __restrict__ t3,
            const __bf16* __restrict__ wt, const float* __restrict__ bias, int Nn) {
    const int lane = threadIdx.x & 31;
    const int wave = threadIdx.x >> 5;        // 8 waves: one 16-col tile each
    const int row0 = blockIdx.x * 16;
    const int colt = wave * 16;
    const int M    = lane & 15;
    const int kb   = (lane >> 4) * 16;        // A: k-base within 32-chunk
    const int kbB  = (lane >> 4) * 16;        // B: k-base within 32-chunk
    const int col  = colt + (lane & 15);
    const float* T[4] = {t0, t1, t2, t3};

    v8f acc;
    const float bv = bias[col];
#pragma unroll
    for (int r = 0; r < 8; ++r) acc[r] = bv;

    int rowA = row0 + M;
    if (rowA >= Nn) rowA = Nn - 1;

#pragma unroll
    for (int t = 0; t < 4; ++t) {
        const float* base = T[t] + (size_t)rowA * HD;
#pragma unroll
        for (int kc = 0; kc < 4; ++kc) {
            const float* ap = base + kc * 32 + kb;
            float4 a0 = *reinterpret_cast<const float4*>(ap);
            float4 a1 = *reinterpret_cast<const float4*>(ap + 4);
            float4 a2 = *reinterpret_cast<const float4*>(ap + 8);
            float4 a3 = *reinterpret_cast<const float4*>(ap + 12);
            v16f af;
            af[0]  = a0.x; af[1]  = a0.y; af[2]  = a0.z; af[3]  = a0.w;
            af[4]  = a1.x; af[5]  = a1.y; af[6]  = a1.z; af[7]  = a1.w;
            af[8]  = a2.x; af[9]  = a2.y; af[10] = a2.z; af[11] = a2.w;
            af[12] = a3.x; af[13] = a3.y; af[14] = a3.z; af[15] = a3.w;
            v16bf A = __builtin_convertvector(af, v16bf);   // packed v_cvt_pk_bf16_f32
            v16bf B = *reinterpret_cast<const v16bf*>(wt + (size_t)t * HD * HD + (size_t)col * HD + kc * 32 + kbB);
            acc = __builtin_amdgcn_wmma_f32_16x16x32_bf16(false, A, false, B, (short)0, acc, false, false);
        }
    }
#pragma unroll
    for (int r = 0; r < 8; ++r) {
        int row = row0 + r + ((lane >> 4) * 8);
        if (row < Nn) z[(size_t)row * HD + col] = acc[r];
    }
}

// ---------- activation + batchnorm ----------
__global__ void k_act(float* z, float alpha, int n) {
    int i = blockIdx.x * blockDim.x + threadIdx.x;
    if (i >= n) return;
    float v = z[i];
    z[i] = v > 0.f ? v : alpha * v;
}

__global__ void k_bn_stats(const float* z, const float* g, const float* be,
                           float* scale, float* shift, int Nn) {
    __shared__ float s1[256], s2[256];
    int c = blockIdx.x;
    float sum = 0.f, sq = 0.f;
    for (int r = threadIdx.x; r < Nn; r += blockDim.x) {
        float v = z[(size_t)r * HD + c];
        sum += v; sq += v * v;
    }
    s1[threadIdx.x] = sum; s2[threadIdx.x] = sq;
    __syncthreads();
    for (int o = 128; o > 0; o >>= 1) {
        if ((int)threadIdx.x < o) { s1[threadIdx.x] += s1[threadIdx.x + o]; s2[threadIdx.x] += s2[threadIdx.x + o]; }
        __syncthreads();
    }
    if (threadIdx.x == 0) {
        float m = s1[0] / (float)Nn;
        float var = s2[0] / (float)Nn - m * m;
        float istd = rsqrtf(var + 1e-5f);
        float sc = g[c] * istd;
        scale[c] = sc;
        shift[c] = be[c] - m * sc;
    }
}

__global__ void k_bn_apply(float* z, const float* scale, const float* shift, int n) {
    int i = blockIdx.x * blockDim.x + threadIdx.x;
    if (i >= n) return;
    int c = i & 127;
    z[i] = z[i] * scale[c] + shift[c];
}

// ---------- fused L2-normalize + final projection H -> 3 ----------
__global__ void k_final(float* out, const float* z, const float* Wm, const float* bm, int Nn) {
    int n = blockIdx.x * blockDim.x + threadIdx.x;
    if (n >= Nn) return;
    const float* row = z + (size_t)n * HD;
    float ss = 0.f;
#pragma unroll 4
    for (int k = 0; k < HD; k += 4) {
        float4 v = *reinterpret_cast<const float4*>(row + k);
        ss += v.x * v.x + v.y * v.y + v.z * v.z + v.w * v.w;
    }
    float inv = 1.f / fmaxf(sqrtf(ss), 1e-12f);
    float a0 = bm[0], a1 = bm[1], a2 = bm[2];
    for (int k = 0; k < HD; ++k) {
        float v = row[k] * inv;
        a0 += v * Wm[k * 3 + 0];
        a1 += v * Wm[k * 3 + 1];
        a2 += v * Wm[k * 3 + 2];
    }
    out[n * 3 + 0] = a0; out[n * 3 + 1] = a1; out[n * 3 + 2] = a2;
}

extern "C" void kernel_launch(void* const* d_in, const int* in_sizes, int n_in,
                              void* d_out, int out_size, void* d_ws, size_t ws_size,
                              hipStream_t stream) {
    const float* x  = (const float*)d_in[0];
    const int*   ei = (const int*)d_in[1];
    const int Nn = in_sizes[0] / 3;
    const int E  = in_sizes[1] / 2;
    const int* src = ei;
    const int* dst = ei + E;
    const float *W1 = (const float*)d_in[2],  *b1 = (const float*)d_in[3];
    const float *W2 = (const float*)d_in[4],  *b2 = (const float*)d_in[5];
    const float *W3 = (const float*)d_in[6],  *b3 = (const float*)d_in[7];
    const float *W4 = (const float*)d_in[8],  *b4 = (const float*)d_in[9];
    const float *g1 = (const float*)d_in[10], *be1 = (const float*)d_in[11];
    const float *g2 = (const float*)d_in[12], *be2 = (const float*)d_in[13];
    const float *g3 = (const float*)d_in[14], *be3 = (const float*)d_in[15];
    const float *Wm = (const float*)d_in[16], *bm  = (const float*)d_in[17];
    float* out = (float*)d_out;

    // carve workspace (256B-aligned slabs)
    uint8_t* p = (uint8_t*)d_ws;
    auto carve = [&](size_t bytes) -> void* {
        void* r = (void*)p;
        p += (bytes + 255) & ~(size_t)255;
        return r;
    };
    float* deg = (float*)carve((size_t)Nn * 4);
    float* wE  = (float*)carve((size_t)E * 4);
    float* t13 = (float*)carve((size_t)Nn * 3 * 4);
    float* t23 = (float*)carve((size_t)Nn * 3 * 4);
    float* t33 = (float*)carve((size_t)Nn * 3 * 4);
    float* T1  = (float*)carve((size_t)Nn * HD * 4);
    float* T2  = (float*)carve((size_t)Nn * HD * 4);
    float* T3  = (float*)carve((size_t)Nn * HD * 4);
    float* H0  = (float*)carve((size_t)Nn * HD * 4);
    float* H1  = (float*)carve((size_t)Nn * HD * 4);
    __bf16* WT = (__bf16*)carve((size_t)4 * HD * HD * 2);
    float* scl = (float*)carve(HD * 4);
    float* shf = (float*)carve(HD * 4);

    const int T = 256;
    auto cdiv = [](long long a, long long b) { return (int)((a + b - 1) / b); };
    const int gE   = cdiv(E, T);
    const int gE32 = cdiv((long long)E * 32, T);
    const int gNH  = cdiv((long long)Nn * HD, T);
    const int gN3  = cdiv((long long)Nn * 3, T);
    const size_t nhBytes = (size_t)Nn * HD * 4;

    // graph normalization
    hipMemsetAsync(deg, 0, (size_t)Nn * 4, stream);
    k_deg<<<gE, T, 0, stream>>>(deg, src, E);
    k_edgew<<<gE, T, 0, stream>>>(wE, deg, src, dst, E);

    // ---- layer 1 (3-wide Chebyshev) ----
    hipMemsetAsync(t13, 0, (size_t)Nn * 3 * 4, stream);
    k_scatter3<<<gE, T, 0, stream>>>(t13, x, wE, src, dst, 1.f, E);
    k_negcopy<<<gN3, T, 0, stream>>>(t23, x, Nn * 3);
    k_scatter3<<<gE, T, 0, stream>>>(t23, t13, wE, src, dst, 2.f, E);
    k_negcopy<<<gN3, T, 0, stream>>>(t33, t13, Nn * 3);
    k_scatter3<<<gE, T, 0, stream>>>(t33, t23, wE, src, dst, 2.f, E);
    k_dense1<<<gNH, T, 0, stream>>>(H0, x, t13, t23, t33, W1, b1, Nn);
    k_act<<<gNH, T, 0, stream>>>(H0, LRELU, Nn * HD);
    k_bn_stats<<<HD, 256, 0, stream>>>(H0, g1, be1, scl, shf, Nn);
    k_bn_apply<<<gNH, T, 0, stream>>>(H0, scl, shf, Nn * HD);

    // fused Chebyshev propagation for 128-wide features: T1,T2,T3 from hin
    auto props = [&](const float* hin) {
        hipMemsetAsync(T1, 0, nhBytes, stream);
        k_scatter128<<<gE32, T, 0, stream>>>(T1, hin, wE, src, dst, 1.f, E);
        k_negcopy<<<gNH, T, 0, stream>>>(T2, hin, Nn * HD);
        k_scatter128<<<gE32, T, 0, stream>>>(T2, T1, wE, src, dst, 2.f, E);
        k_negcopy<<<gNH, T, 0, stream>>>(T3, T1, Nn * HD);
        k_scatter128<<<gE32, T, 0, stream>>>(T3, T2, wE, src, dst, 2.f, E);
    };
    const int gW = cdiv(Nn, 16);

    // ---- layer 2 ----
    props(H0);
    k_twei<<<cdiv(4 * HD * HD, T), T, 0, stream>>>(WT, W2);
    k_cheb_wmma<<<gW, 256, 0, stream>>>(H1, H0, T1, T2, T3, WT, b2, Nn);
    k_act<<<gNH, T, 0, stream>>>(H1, LRELU, Nn * HD);
    k_bn_stats<<<HD, 256, 0, stream>>>(H1, g2, be2, scl, shf, Nn);
    k_bn_apply<<<gNH, T, 0, stream>>>(H1, scl, shf, Nn * HD);

    // ---- layer 3 ----
    props(H1);
    k_twei<<<cdiv(4 * HD * HD, T), T, 0, stream>>>(WT, W3);
    k_cheb_wmma<<<gW, 256, 0, stream>>>(H0, H1, T1, T2, T3, WT, b3, Nn);
    k_act<<<gNH, T, 0, stream>>>(H0, 0.0f, Nn * HD);   // ReLU
    k_bn_stats<<<HD, 256, 0, stream>>>(H0, g3, be3, scl, shf, Nn);
    k_bn_apply<<<gNH, T, 0, stream>>>(H0, scl, shf, Nn * HD);

    // ---- layer 4 + normalize + projection ----
    props(H0);
    k_twei<<<cdiv(4 * HD * HD, T), T, 0, stream>>>(WT, W4);
    k_cheb_wmma<<<gW, 256, 0, stream>>>(H1, H0, T1, T2, T3, WT, b4, Nn);
    k_final<<<cdiv(Nn, T), T, 0, stream>>>(out, H1, Wm, bm, Nn);
}
